// RooDecAttention_56272661512620
// MI455X (gfx1250) — matrix operand order repaired and checked by reading, not compiled, not run on hardware
//
#include <hip/hip_runtime.h>

// ---------------------------------------------------------------------------
// RooDec attention for MI455X (gfx1250, wave32, WMMA).
// f16 data / f32 accumulate via v_wmma_f32_16x16x32_f16.
// GEMM B-operand staged through LDS with async global->LDS copies (ASYNCcnt),
// double-buffered and overlapped with WMMA.
// ---------------------------------------------------------------------------

typedef __attribute__((ext_vector_type(16))) _Float16 v16h;
typedef __attribute__((ext_vector_type(8)))  float    v8f;
typedef __attribute__((ext_vector_type(8)))  unsigned v8u;

namespace cfg {
constexpr int B  = 8;
constexpr int S  = 1024;
constexpr int D  = 1024;
constexpr int DA = 256;
constexpr int H  = 16;
constexpr int R  = 32;     // sqrt(S): 32 blocks of 32 tokens
constexpr int HD = 64;     // head dim
}

// ---------------------------------------------------------------------------
// WMMA fragment loaders (dword units = packed f16 K-pairs), K order chosen here (same for A and B):
//   A (16x32): m = lane&15 ; VGPR v holds K-pair v + 8*(lane>>4)
//   B (32x16): n = lane&15 ; VGPR v holds K-pair v + 8*(lane>>4)
//   C/D (16x16 f32): VGPR v -> M = v + 8*(lane>>4), N = lane&15
// ---------------------------------------------------------------------------

// A tile: Ap = dword ptr at [m0, k0/2]; ld_dw = row stride in dwords.
__device__ __forceinline__ v16h load_a_u(const unsigned* __restrict__ Ap, int ld_dw, int lane) {
  v8u t;
  const int hf = lane >> 4, m = lane & 15;
  const unsigned* r = Ap + (size_t)m * ld_dw;
  #pragma unroll
  for (int v = 0; v < 8; ++v) {
    const int p = v + 8 * hf;
    t[v] = r[p];
  }
  return __builtin_bit_cast(v16h, t);
}

// Packed-B tile from global: Bp = dword ptr at [k0/2][n0] of [K/2][ld] dwords.
__device__ __forceinline__ v16h load_bp_u(const unsigned* __restrict__ Bp, int ld, int lane) {
  v8u t;
  const int hf = lane >> 4, n = lane & 15;
  #pragma unroll
  for (int v = 0; v < 8; ++v)
    t[v] = Bp[(size_t)(v + 8 * hf) * ld + n];
  return __builtin_bit_cast(v16h, t);
}

// Packed-B tile from an LDS stage of shape [16][128] dwords.
__device__ __forceinline__ v16h load_bp_lds(const unsigned* Lp, int lane) {
  v8u t;
  const int hf = lane >> 4, n = lane & 15;
  #pragma unroll
  for (int v = 0; v < 8; ++v)
    t[v] = Lp[(v + 8 * hf) * 128 + n];
  return __builtin_bit_cast(v16h, t);
}

// Transposed-operand B (K^T): logical B[k][n] = mem[n][k]; Kp = dword ptr at
// [n0 row, k0/2]; ld_dw = row stride in dwords. K-pairs contiguous -> 1 dword.
__device__ __forceinline__ v16h load_bT_u(const unsigned* __restrict__ Kp, int ld_dw, int lane) {
  v8u t;
  const int hf = lane >> 4, n = lane & 15;
  const unsigned* r = Kp + (size_t)n * ld_dw;
  #pragma unroll
  for (int v = 0; v < 8; ++v)
    t[v] = r[v + 8 * hf];
  return __builtin_bit_cast(v16h, t);
}

__device__ __forceinline__ unsigned pack2(float lo, float hi) {
  const _Float16 l = (_Float16)lo, h = (_Float16)hi;
  return (unsigned)__builtin_bit_cast(unsigned short, l) |
         ((unsigned)__builtin_bit_cast(unsigned short, h) << 16);
}

// ---------------------------------------------------------------------------
// Async global->LDS copy of one 16x128-dword B tile (8 KB) by 256 threads:
// 2 passes x b128 per thread. Tracked by ASYNCcnt (ISA 15.18 / 5.7.2).
// LDS byte offset = low 32 bits of the generic pointer (LDS aperture, ISA 10.2).
// ---------------------------------------------------------------------------
__device__ __forceinline__ void async_copy_b_tile(const unsigned* gsrc, int ldN,
                                                  unsigned* ldst, int tid) {
  const int r  = tid >> 5;          // 0..7
  const int c4 = (tid & 31) << 2;   // dword column, 16B aligned
  #pragma unroll
  for (int pass = 0; pass < 2; ++pass) {
    const unsigned* g = gsrc + (size_t)(r + 8 * pass) * ldN + c4;
    const unsigned loff = (unsigned)(uintptr_t)(ldst + (r + 8 * pass) * 128 + c4);
    asm volatile("global_load_async_to_lds_b128 %0, %1, off"
                 :: "v"(loff), "v"(g)
                 : "memory");
  }
}

// ---------------------------------------------------------------------------
// fp32 -> f16 elementwise (for x: A-operand + residual source stays fp32)
// ---------------------------------------------------------------------------
__global__ void cvt_f32_f16(const float* __restrict__ src, _Float16* __restrict__ dst, int n) {
  int i = blockIdx.x * blockDim.x + threadIdx.x;
  if (i < n) dst[i] = (_Float16)src[i];
}

// fp32 [K,N] -> packed f16 dwords [K/2][N] (row-pair per dword, even row low)
__global__ void cvt_pack_f32(const float* __restrict__ src, unsigned* __restrict__ dst,
                             int Kp, int N) {
  int i = blockIdx.x * blockDim.x + threadIdx.x;    // over Kp*N
  if (i >= Kp * N) return;
  const int p = i / N, n = i % N;
  const float lo = src[(size_t)(2 * p) * N + n];
  const float hi = src[(size_t)(2 * p + 1) * N + n];
  dst[i] = pack2(lo, hi);
}

// ---------------------------------------------------------------------------
// Block pooling: pooled[b,r,d] = mean_{t<32} x[b, r*32+t, d]   (f16 out)
// ---------------------------------------------------------------------------
__global__ void pool_kernel(const float* __restrict__ x, _Float16* __restrict__ pooled) {
  using namespace cfg;
  int i = blockIdx.x * blockDim.x + threadIdx.x;    // over B*R*D
  if (i >= B * R * D) return;
  const int d = i % D;
  const int br = i / D;
  const int b = br / R, r = br % R;
  const float* base = x + ((size_t)(b * S + r * 32)) * D + d;
  float s = 0.f;
  #pragma unroll
  for (int t = 0; t < 32; ++t) s += base[(size_t)t * D];
  pooled[i] = (_Float16)(s * (1.f / 32.f));
}

// ---------------------------------------------------------------------------
// WMMA GEMM: C[M,N] = A[M,K] @ B[K,N]; A f16 row-major, B packed dwords.
// Each wave computes a 32x32 patch (2x2 tiles); block = 8 waves = 64x128.
// B tiles flow global -> LDS via async copies, double-buffered; all 8 waves
// share the staged tile. Grids must tile M,N exactly (barriers in loop).
// ---------------------------------------------------------------------------
enum { OUT_F32 = 0, OUT_F32_RESID = 1, OUT_F16 = 2, OUT_PK = 3 };

template <int MODE>
__global__ __launch_bounds__(256) void gemm_f16_wmma(
    const _Float16* __restrict__ A, const unsigned* __restrict__ Bpk,
    const float* __restrict__ resid, float* __restrict__ outF,
    _Float16* __restrict__ outH, unsigned* __restrict__ outPk,
    int M, int N, int K) {
  __shared__ __align__(16) unsigned btile[2][16 * 128];   // 2 x 8 KB stage

  const int tid  = threadIdx.x;
  const int lane = tid & 31;
  const int w    = tid >> 5;
  const int m0 = blockIdx.y * 64 + (w >> 2) * 32;
  const int n0 = blockIdx.x * 128 + (w & 3) * 32;
  const int hf = lane >> 4, nn = lane & 15;

  const int ld_dw = K >> 1;
  const unsigned* A0 = (const unsigned*)A + (size_t)m0 * ld_dw;
  const unsigned* A1 = A0 + (size_t)16 * ld_dw;
  const unsigned* Bblk = Bpk + blockIdx.x * 128;          // column panel base

  // prologue: stage first B tile
  async_copy_b_tile(Bblk, N, &btile[0][0], tid);

  v8f c00 = {}, c01 = {}, c10 = {}, c11 = {};
  for (int k0 = 0; k0 < K; k0 += 32) {
    const int buf = (k0 >> 5) & 1;
    if (k0 + 32 < K) {          // issue next tile, then wait for current only
      async_copy_b_tile(Bblk + (size_t)((k0 + 32) >> 1) * N, N,
                        &btile[buf ^ 1][0], tid);
      asm volatile("s_wait_asynccnt 0x2" ::: "memory");
      __builtin_prefetch(A0 + (size_t)(lane & 15) * ld_dw + ((k0 + 32) >> 1), 0, 0);
    } else {
      asm volatile("s_wait_asynccnt 0x0" ::: "memory");
    }
    __syncthreads();            // staged tile visible to all 8 waves

    v16h a0 = load_a_u(A0 + (k0 >> 1), ld_dw, lane);
    v16h a1 = load_a_u(A1 + (k0 >> 1), ld_dw, lane);
    const unsigned* Lb = &btile[buf][0] + (w & 3) * 32;
    v16h b0 = load_bp_lds(Lb, lane);
    v16h b1 = load_bp_lds(Lb + 16, lane);

    c00 = __builtin_amdgcn_wmma_f32_16x16x32_f16(false, a0, false, b0, (short)0, c00, false, false);
    c01 = __builtin_amdgcn_wmma_f32_16x16x32_f16(false, a0, false, b1, (short)0, c01, false, false);
    c10 = __builtin_amdgcn_wmma_f32_16x16x32_f16(false, a1, false, b0, (short)0, c10, false, false);
    c11 = __builtin_amdgcn_wmma_f32_16x16x32_f16(false, a1, false, b1, (short)0, c11, false, false);

    __syncthreads();            // all reads done before buffer is re-staged
  }

  if constexpr (MODE == OUT_F32 || MODE == OUT_F32_RESID) {
    auto st = [&](const v8f& c, int mt, int nt) {
      #pragma unroll
      for (int v = 0; v < 8; ++v) {
        const size_t idx = (size_t)(mt + v + 8 * hf) * N + nt + nn;
        float val = c[v];
        if constexpr (MODE == OUT_F32_RESID) val += resid[idx];
        outF[idx] = val;
      }
    };
    st(c00, m0, n0); st(c01, m0, n0 + 16); st(c10, m0 + 16, n0); st(c11, m0 + 16, n0 + 16);
  } else if constexpr (MODE == OUT_F16) {
    auto st = [&](const v8f& c, int mt, int nt) {
      #pragma unroll
      for (int v = 0; v < 8; ++v)
        outH[(size_t)(mt + v + 8 * hf) * N + nt + nn] = (_Float16)c[v];
    };
    st(c00, m0, n0); st(c01, m0, n0 + 16); st(c10, m0 + 16, n0); st(c11, m0 + 16, n0 + 16);
  } else {  // OUT_PK: adjacent rows live in adjacent C-VGPRs -> free packing
    auto st = [&](const v8f& c, int mt, int nt) {
      #pragma unroll
      for (int v = 0; v < 8; v += 2) {
        const int row = mt + v + 8 * hf;
        outPk[(size_t)(row >> 1) * N + nt + nn] = pack2(c[v], c[v + 1]);
      }
    };
    st(c00, m0, n0); st(c01, m0, n0 + 16); st(c10, m0 + 16, n0); st(c11, m0 + 16, n0 + 16);
  }
}

// ---------------------------------------------------------------------------
// Block selection: one wave per (b,s); lane == block r (R == 32 == wave32).
// logits = qsel . ksel / 16 ; softmax over lanes ; bit r set if selected.
// ---------------------------------------------------------------------------
__global__ __launch_bounds__(256) void select_kernel(
    const float* __restrict__ qsel, const float* __restrict__ ksel,
    unsigned* __restrict__ maskbits) {
  using namespace cfg;
  const int wid = blockIdx.x * (blockDim.x >> 5) + (threadIdx.x >> 5);
  if (wid >= B * S) return;
  const int lane = threadIdx.x & 31;
  const int b = wid / S, s = wid % S;

  const float4* q4 = (const float4*)(qsel + (size_t)wid * DA);
  const float4* k4 = (const float4*)(ksel + ((size_t)b * R + lane) * DA);
  float acc = 0.f;
  #pragma unroll 4
  for (int d = 0; d < DA / 4; ++d) {
    const float4 qv = q4[d], kv = k4[d];
    acc = fmaf(qv.x, kv.x, acc);
    acc = fmaf(qv.y, kv.y, acc);
    acc = fmaf(qv.z, kv.z, acc);
    acc = fmaf(qv.w, kv.w, acc);
  }
  acc *= (1.f / 16.f);                           // 1/sqrt(DA)

  float mx = acc;
  #pragma unroll
  for (int o = 16; o; o >>= 1) mx = fmaxf(mx, __shfl_xor(mx, o, 32));
  const float e = __expf(acc - mx);
  float sum = e;
  #pragma unroll
  for (int o = 16; o; o >>= 1) sum += __shfl_xor(sum, o, 32);
  const float prob = e / sum;

  const bool sel = (prob >= 0.5f) || ((s >> 5) == lane);
  unsigned bit = sel ? (1u << lane) : 0u;
  #pragma unroll
  for (int o = 16; o; o >>= 1) bit |= __shfl_xor(bit, o, 32);
  if (lane == 0) maskbits[wid] = bit;
}

// ---------------------------------------------------------------------------
// Flash-style masked MHA. One wave per (b, h, 16-row q tile).
// Key chunk = 32 keys = exactly one selection block -> mask is one bit/row.
// QK^T: 4 WMMAs / chunk ; PV: 4 WMMAs / chunk ; P transposed through LDS.
// ---------------------------------------------------------------------------
__global__ __launch_bounds__(128) void attn_kernel(
    const _Float16* __restrict__ qh, const _Float16* __restrict__ kh,
    const unsigned* __restrict__ vpk, const unsigned* __restrict__ maskbits,
    _Float16* __restrict__ ctx) {
  using namespace cfg;
  __shared__ __align__(16) _Float16 ldsP[4][16 * 32];   // per-wave P tile

  const int lane = threadIdx.x & 31;
  const int w    = threadIdx.x >> 5;
  const int hf = lane >> 4, ln = lane & 15;
  const int qt = blockIdx.x * 4 + w;
  const int h  = blockIdx.y;
  const int b  = blockIdx.z;
  const int q0 = qt * 16;

  const size_t base = (size_t)b * S * D + (size_t)h * HD;
  const int Ddw = D >> 1;

  // Q A-fragments: 16x64 tile -> two 16x32 fragments, resident for whole loop.
  const unsigned* Qp = (const unsigned*)(qh + base + (size_t)q0 * D);
  const v16h qa0 = load_a_u(Qp, Ddw, lane);
  const v16h qa1 = load_a_u(Qp + 16, Ddw, lane);

  // Selection bitwords: lanes 0..15 (mirrored 16..31) hold row ln's word.
  const unsigned mw = maskbits[(size_t)b * S + q0 + ln];

  v8f acc0 = {}, acc1 = {}, acc2 = {}, acc3 = {};
  float mrow[8], lrow[8];
  #pragma unroll
  for (int v = 0; v < 8; ++v) { mrow[v] = -1e30f; lrow[v] = 0.f; }

  for (int kc = 0; kc < R; ++kc) {               // 32 chunks of 32 keys
    const int key0 = kc * 32;

    // ---- scores: S[16 q x 32 k] as two 16x16 f32 fragments -----------------
    v8f s0 = {}, s1 = {};
    {
      const unsigned* Kp = (const unsigned*)(kh + base + (size_t)key0 * D);
      v16h b0 = load_bT_u(Kp, Ddw, lane);                 // keys +0..15,  dims 0..31
      v16h b1 = load_bT_u(Kp + 16, Ddw, lane);            // dims 32..63
      s0 = __builtin_amdgcn_wmma_f32_16x16x32_f16(false, qa0, false, b0, (short)0, s0, false, false);
      s0 = __builtin_amdgcn_wmma_f32_16x16x32_f16(false, qa1, false, b1, (short)0, s0, false, false);
      const unsigned* Kp2 = Kp + (size_t)16 * Ddw;        // keys +16..31
      v16h b2 = load_bT_u(Kp2, Ddw, lane);
      v16h b3 = load_bT_u(Kp2 + 16, Ddw, lane);
      s1 = __builtin_amdgcn_wmma_f32_16x16x32_f16(false, qa0, false, b2, (short)0, s1, false, false);
      s1 = __builtin_amdgcn_wmma_f32_16x16x32_f16(false, qa1, false, b3, (short)0, s1, false, false);
      if (kc + 1 < R)
        __builtin_prefetch(kh + base + (size_t)(key0 + 32) * D + lane * 4, 0, 0);
    }

    // ---- mask + online softmax over this 32-key block ----------------------
    #pragma unroll
    for (int v = 0; v < 8; ++v) {
      const int row = v + 8 * hf;
      const unsigned wvord = __shfl(mw, row, 32); // row's selection bitword
      const bool ok = (wvord >> kc) & 1u;
      float x0 = ok ? s0[v] * 0.125f : -1e30f;    // 1/sqrt(HD)
      float x1 = ok ? s1[v] * 0.125f : -1e30f;

      float m2 = fmaxf(x0, x1);                   // row-max across 16 lanes/half
      #pragma unroll
      for (int o = 1; o < 16; o <<= 1) m2 = fmaxf(m2, __shfl_xor(m2, o, 32));
      const float mnew = fmaxf(mrow[v], m2);
      const float resc = __expf(mrow[v] - mnew);
      const float p0 = __expf(x0 - mnew);
      const float p1 = __expf(x1 - mnew);
      float rs = p0 + p1;
      #pragma unroll
      for (int o = 1; o < 16; o <<= 1) rs += __shfl_xor(rs, o, 32);
      lrow[v] = lrow[v] * resc + rs;
      mrow[v] = mnew;
      acc0[v] = acc0[v] * resc; acc1[v] = acc1[v] * resc;
      acc2[v] = acc2[v] * resc; acc3[v] = acc3[v] * resc;

      // stash P (C-layout) into LDS for re-read in A-layout
      ldsP[w][row * 32 + ln]      = (_Float16)p0;
      ldsP[w][row * 32 + 16 + ln] = (_Float16)p1;
    }

    // ---- PV: P[16x32] @ V[32x64] -> 4 WMMAs --------------------------------
    const v16h pa = load_a_u((const unsigned*)&ldsP[w][0], 16, lane);
    const unsigned* Vb = vpk + (size_t)(((size_t)b * S + key0) >> 1) * D + h * HD;
    v16h bv0 = load_bp_u(Vb,      D, lane);
    v16h bv1 = load_bp_u(Vb + 16, D, lane);
    v16h bv2 = load_bp_u(Vb + 32, D, lane);
    v16h bv3 = load_bp_u(Vb + 48, D, lane);
    acc0 = __builtin_amdgcn_wmma_f32_16x16x32_f16(false, pa, false, bv0, (short)0, acc0, false, false);
    acc1 = __builtin_amdgcn_wmma_f32_16x16x32_f16(false, pa, false, bv1, (short)0, acc1, false, false);
    acc2 = __builtin_amdgcn_wmma_f32_16x16x32_f16(false, pa, false, bv2, (short)0, acc2, false, false);
    acc3 = __builtin_amdgcn_wmma_f32_16x16x32_f16(false, pa, false, bv3, (short)0, acc3, false, false);
  }

  // ---- epilogue: normalize, store ctx (f16) --------------------------------
  _Float16* Cp = ctx + base + (size_t)q0 * D;
  #pragma unroll
  for (int v = 0; v < 8; ++v) {
    const int row = v + 8 * hf;
    const float inv = 1.f / lrow[v];
    Cp[(size_t)row * D + ln]      = (_Float16)(acc0[v] * inv);
    Cp[(size_t)row * D + 16 + ln] = (_Float16)(acc1[v] * inv);
    Cp[(size_t)row * D + 32 + ln] = (_Float16)(acc2[v] * inv);
    Cp[(size_t)row * D + 48 + ln] = (_Float16)(acc3[v] * inv);
  }
}

// ---------------------------------------------------------------------------
// Host launch
// ---------------------------------------------------------------------------
extern "C" void kernel_launch(void* const* d_in, const int* in_sizes, int n_in,
                              void* d_out, int out_size, void* d_ws, size_t ws_size,
                              hipStream_t stream) {
  using namespace cfg;
  (void)in_sizes; (void)n_in; (void)out_size; (void)ws_size;

  const float* x     = (const float*)d_in[0];   // [B,S,D]
  const float* Wu    = (const float*)d_in[1];   // [D,D]
  const float* Wksel = (const float*)d_in[2];   // [D,DA]
  const float* Wqsel = (const float*)d_in[3];   // [D,DA]
  const float* Wq    = (const float*)d_in[4];   // [D,D]
  const float* Wk    = (const float*)d_in[5];   // [D,D]
  const float* Wv    = (const float*)d_in[6];   // [D,D]
  const float* Wo    = (const float*)d_in[7];   // [D,D]
  float* out = (float*)d_out;                   // [B,S,D]

  // ---- workspace carve (256B aligned) ----
  size_t off = 0;
  auto carve = [&](size_t bytes) -> void* {
    void* p = (char*)d_ws + off;
    off += (bytes + 255) & ~(size_t)255;
    return p;
  };
  const int nX  = B * S * D;        // 8388608
  const int nW  = D * D;            // 1048576
  const int nWs = D * DA;           // 262144
  const int nP  = B * R * D;        // 262144
  _Float16* x_h      = (_Float16*)carve((size_t)nX * 2);
  unsigned* Wu_pk    = (unsigned*)carve((size_t)nW * 2);     // [D/2][D] dwords
  unsigned* Wksel_pk = (unsigned*)carve((size_t)nWs * 2);
  unsigned* Wqsel_pk = (unsigned*)carve((size_t)nWs * 2);
  unsigned* Wq_pk    = (unsigned*)carve((size_t)nW * 2);
  unsigned* Wk_pk    = (unsigned*)carve((size_t)nW * 2);
  unsigned* Wv_pk    = (unsigned*)carve((size_t)nW * 2);
  unsigned* Wo_pk    = (unsigned*)carve((size_t)nW * 2);
  _Float16* pooled_h = (_Float16*)carve((size_t)nP * 2);
  _Float16* root_h   = (_Float16*)carve((size_t)nP * 2);
  float*    ksel_f   = (float*)carve((size_t)B * R * DA * 4);
  float*    qsel_f   = (float*)carve((size_t)B * S * DA * 4);
  _Float16* q_h      = (_Float16*)carve((size_t)nX * 2);
  _Float16* k_h      = (_Float16*)carve((size_t)nX * 2);
  unsigned* v_pk     = (unsigned*)carve((size_t)nX * 2);     // [B*S/2][D] dwords
  _Float16* ctx_h    = (_Float16*)carve((size_t)nX * 2);
  unsigned* maskbits = (unsigned*)carve((size_t)B * S * 4);

  // ---- conversions: x elementwise; weights into packed-pair dword layout ----
  cvt_f32_f16<<<(nX + 255) / 256, 256, 0, stream>>>(x, x_h, nX);
  auto cvtpk = [&](const float* s, unsigned* d, int K, int N) {
    const int n = (K / 2) * N;
    cvt_pack_f32<<<(n + 255) / 256, 256, 0, stream>>>(s, d, K / 2, N);
  };
  cvtpk(Wu, Wu_pk, D, D);
  cvtpk(Wksel, Wksel_pk, D, DA);
  cvtpk(Wqsel, Wqsel_pk, D, DA);
  cvtpk(Wq, Wq_pk, D, D);
  cvtpk(Wk, Wk_pk, D, D);
  cvtpk(Wv, Wv_pk, D, D);
  cvtpk(Wo, Wo_pk, D, D);

  // ---- block pooling ----
  pool_kernel<<<(nP + 255) / 256, 256, 0, stream>>>(x, pooled_h);

  // ---- root_emb = pooled @ Wu  [256 x 1024], f16 out ----
  gemm_f16_wmma<OUT_F16><<<dim3(D / 128, (B * R) / 64), 256, 0, stream>>>(
      pooled_h, Wu_pk, nullptr, nullptr, root_h, nullptr, B * R, D, D);

  // ---- ksel = root_emb @ Wk_sel  [256 x 256], f32 out ----
  gemm_f16_wmma<OUT_F32><<<dim3(DA / 128, (B * R) / 64), 256, 0, stream>>>(
      root_h, Wksel_pk, nullptr, ksel_f, nullptr, nullptr, B * R, DA, D);

  // ---- qsel = x @ Wq_sel  [8192 x 256], f32 out ----
  gemm_f16_wmma<OUT_F32><<<dim3(DA / 128, (B * S) / 64), 256, 0, stream>>>(
      x_h, Wqsel_pk, nullptr, qsel_f, nullptr, nullptr, B * S, DA, D);

  // ---- block selection -> per-token 32-bit mask ----
  select_kernel<<<(B * S) / 8, 256, 0, stream>>>(qsel_f, ksel_f, maskbits);

  // ---- q/k/v projections  [8192 x 1024]; v written pair-packed ----
  gemm_f16_wmma<OUT_F16><<<dim3(D / 128, (B * S) / 64), 256, 0, stream>>>(
      x_h, Wq_pk, nullptr, nullptr, q_h, nullptr, B * S, D, D);
  gemm_f16_wmma<OUT_F16><<<dim3(D / 128, (B * S) / 64), 256, 0, stream>>>(
      x_h, Wk_pk, nullptr, nullptr, k_h, nullptr, B * S, D, D);
  gemm_f16_wmma<OUT_PK><<<dim3(D / 128, (B * S) / 64), 256, 0, stream>>>(
      x_h, Wv_pk, nullptr, nullptr, nullptr, v_pk, B * S, D, D);

  // ---- masked flash attention ----
  attn_kernel<<<dim3((S / 16) / 4, H, B), 128, 0, stream>>>(
      q_h, k_h, v_pk, maskbits, ctx_h);

  // ---- out = ctx @ Wo + x  (f32 residual, f32 output) ----
  gemm_f16_wmma<OUT_F32_RESID><<<dim3(D / 128, (B * S) / 64), 256, 0, stream>>>(
      ctx_h, Wo_pk, x, out, nullptr, nullptr, B * S, D, D);
}
